// QuerySAT_27144193311178
// MI455X (gfx1250) — matrix operand order repaired and checked by reading, not compiled, not run on hardware
//
#include <hip/hip_runtime.h>
#include <math.h>
#include <stdint.h>

// ============================================================================
// QuerySAT-style GNN rounds for MI455X (gfx1250, wave32, WMMA).
// MLPs run on v_wmma_f32_16x16x32_bf16 (bf16 in / f32 accumulate).
// Weights are pre-swizzled once into WMMA-B fragment-major layout so the
// inner loop does ONE 32B vector load per B fragment (2x global_load_b128)
// instead of 16 strided global_load_u16 + v_mov packing.
// Segment sums run as per-edge 128-lane float atomics.
// ============================================================================

typedef __bf16 bf16_t;
typedef __attribute__((ext_vector_type(16))) __bf16 v16bf;
typedef __attribute__((ext_vector_type(8)))  float  v8f;

// ---------- WMMA A fragment from LDS (wave32, contiguous k order, same as k_packB) ------
// A (16x32 bf16), row-major in LDS with leading dim lda:
//   lane L: M = L&15, kh = L>>4; regs 0-3 hold K = kh*16 + 0..7,
//   regs 4-7 hold K = kh*16 + 8 + 0..7 (2 elems per 32-bit reg).
__device__ __forceinline__ v16bf load_a_frag(const bf16_t* __restrict__ As,
                                             int lda, int m, int kh, int k0) {
  const bf16_t* r = As + m * lda + k0;
  v16bf a;
#pragma unroll
  for (int i = 0; i < 8; ++i) a[i] = r[kh * 16 + i];
#pragma unroll
  for (int i = 0; i < 8; ++i) a[8 + i] = r[kh * 16 + 8 + i];
  return a;
}

// One dense layer on a 16-row tile: Y[16][DOUT] = act(As[16][DIN] @ W + b).
// 8 waves per block; wave w covers column tiles w, w+8, ...
// Wsw is fragment-major: fragment (kt,nt) at ((kt*NT)+nt)*512 bf16,
// lane L's 16 elements contiguous at lane*16.
// D layout: reg r -> row kh*8+r, col = lane&15 (+n0).
template <int DIN, int DOUT, bool RELU>
__device__ __forceinline__ void layer_bf16(const bf16_t* __restrict__ As,
                                           const bf16_t* __restrict__ Wsw,
                                           const float* __restrict__ bias,
                                           bf16_t* __restrict__ Hout,  // bf16 LDS out (hidden)
                                           float* __restrict__ Fout) { // f32 LDS out (final)
  constexpr int NT = DOUT / 16;
  constexpr int KT = DIN / 32;
  const int lane = threadIdx.x & 31;
  const int wave = threadIdx.x >> 5;
  const int nl   = lane & 15;
  const int kh   = lane >> 4;
  const int mb   = kh * 8;
  for (int t = wave; t < NT; t += 8) {
    const int n0 = t * 16;
    const bf16_t* wp = Wsw + ((size_t)t << 9) + (lane << 4);
    v8f acc = {};
#pragma unroll
    for (int kt = 0; kt < KT; ++kt) {
      v16bf a = load_a_frag(As, DIN, nl, kh, kt * 32);
      v16bf b = *(const v16bf*)(wp + (size_t)kt * NT * 512);
      acc = __builtin_amdgcn_wmma_f32_16x16x32_bf16(false, a, false, b,
                                                    (short)0, acc, false, false);
    }
    const float bv = bias[n0 + nl];
#pragma unroll
    for (int r = 0; r < 8; ++r) {
      float v = acc[r] + bv;
      if (RELU) v = v > 0.f ? v : 0.f;
      if (Hout) Hout[(mb + r) * DOUT + n0 + nl] = (bf16_t)v;
      else      Fout[(mb + r) * DOUT + n0 + nl] = v;
    }
  }
}

__device__ __forceinline__ void pack4(bf16_t* __restrict__ d, float4 v) {
  d[0] = (bf16_t)v.x; d[1] = (bf16_t)v.y; d[2] = (bf16_t)v.z; d[3] = (bf16_t)v.w;
}

// ---------------------------- q_mlp: 320->128->128->128 ---------------------
__global__ __launch_bounds__(256) void k_qmlp(
    const float* __restrict__ l, const float* __restrict__ noise_r,
    const bf16_t* __restrict__ W1, const float* __restrict__ b1,
    const bf16_t* __restrict__ W2, const float* __restrict__ b2,
    const bf16_t* __restrict__ W3, const float* __restrict__ b3,
    float* __restrict__ qout, int V) {
  __shared__ bf16_t As[16 * 320];
  __shared__ bf16_t Hs1[16 * 128];
  __shared__ bf16_t Hs2[16 * 128];
  __shared__ float  Fs[16 * 128];
  const int row0 = blockIdx.x * 16;
  for (int idx = threadIdx.x; idx < 16 * 320 / 4; idx += 256) {
    const int e = idx * 4;
    const int r = e / 320, c = e % 320;
    const size_t rr = (size_t)(row0 + r);
    float4 v;
    if (c < 128)      v = *(const float4*)&l[rr * 128 + c];
    else if (c < 256) v = *(const float4*)&l[(size_t)(V + row0 + r) * 128 + (c - 128)];
    else              v = *(const float4*)&noise_r[rr * 64 + (c - 256)];
    pack4(&As[r * 320 + c], v);
  }
  __syncthreads();
  layer_bf16<320, 128, true >(As,  W1, b1, Hs1, nullptr);
  __syncthreads();
  layer_bf16<128, 128, true >(Hs1, W2, b2, Hs2, nullptr);
  __syncthreads();
  layer_bf16<128, 128, false>(Hs2, W3, b3, nullptr, Fs);
  __syncthreads();
  for (int idx = threadIdx.x; idx < 16 * 128 / 4; idx += 256) {
    const int e = idx * 4;
    const int r = e >> 7, c = e & 127;
    *(float4*)&qout[(size_t)(row0 + r) * 128 + c] = *(const float4*)&Fs[e];
  }
}

// --------------------------- v_update: 512->256->256->256 -------------------
__global__ __launch_bounds__(256) void k_vupdate(
    float* __restrict__ l,
    const float* __restrict__ pc2l, const float* __restrict__ nc2l,
    const bf16_t* __restrict__ W1, const float* __restrict__ b1,
    const bf16_t* __restrict__ W2, const float* __restrict__ b2,
    const bf16_t* __restrict__ W3, const float* __restrict__ b3, int V) {
  __shared__ bf16_t As[16 * 512];
  __shared__ bf16_t Hs1[16 * 256];
  __shared__ bf16_t Hs2[16 * 256];
  __shared__ float  Fs[16 * 256];
  const int row0 = blockIdx.x * 16;
  for (int idx = threadIdx.x; idx < 16 * 512 / 4; idx += 256) {
    const int e = idx * 4;
    const int r = e / 512, c = e % 512;
    const size_t rr = (size_t)(row0 + r);
    float4 v;
    if (c < 128)      v = *(const float4*)&l[rr * 128 + c];
    else if (c < 256) v = *(const float4*)&l[(size_t)(V + row0 + r) * 128 + (c - 128)];
    else if (c < 384) v = *(const float4*)&pc2l[rr * 128 + (c - 256)];
    else              v = *(const float4*)&nc2l[rr * 128 + (c - 384)];
    pack4(&As[r * 512 + c], v);
  }
  __syncthreads();
  layer_bf16<512, 256, true >(As,  W1, b1, Hs1, nullptr);
  __syncthreads();
  layer_bf16<256, 256, true >(Hs1, W2, b2, Hs2, nullptr);
  __syncthreads();
  layer_bf16<256, 256, false>(Hs2, W3, b3, nullptr, Fs);
  __syncthreads();
  // l_new = concat(v_new[:, :128], v_new[:, 128:], axis=0), written in place
  for (int idx = threadIdx.x; idx < 16 * 256 / 4; idx += 256) {
    const int e = idx * 4;
    const int r = e >> 8, c = e & 255;
    const float4 v = *(const float4*)&Fs[e];
    if (c < 128) *(float4*)&l[(size_t)(row0 + r) * 128 + c]             = v;
    else         *(float4*)&l[(size_t)(V + row0 + r) * 128 + (c - 128)] = v;
  }
}

// --------------------------- c_update: 384->128->128->128 -------------------
__global__ __launch_bounds__(256) void k_cupdate(
    float* __restrict__ c,
    const float* __restrict__ l2c, const float* __restrict__ q2c,
    const bf16_t* __restrict__ W1, const float* __restrict__ b1,
    const bf16_t* __restrict__ W2, const float* __restrict__ b2,
    const bf16_t* __restrict__ W3, const float* __restrict__ b3, int C) {
  __shared__ bf16_t As[16 * 384];
  __shared__ bf16_t Hs1[16 * 128];
  __shared__ bf16_t Hs2[16 * 128];
  __shared__ float  Fs[16 * 128];
  const int row0 = blockIdx.x * 16;
  for (int idx = threadIdx.x; idx < 16 * 384 / 4; idx += 256) {
    const int e = idx * 4;
    const int r = e / 384, cc = e % 384;
    const size_t rr = (size_t)(row0 + r);
    float4 v;
    if (cc < 128)      v = *(const float4*)&l2c[rr * 128 + cc];
    else if (cc < 256) v = *(const float4*)&c[rr * 128 + (cc - 128)];
    else {                                              // e_emb = exp(-q2c)
      v = *(const float4*)&q2c[rr * 128 + (cc - 256)];
      v.x = __expf(-v.x); v.y = __expf(-v.y); v.z = __expf(-v.z); v.w = __expf(-v.w);
    }
    pack4(&As[r * 384 + cc], v);
  }
  __syncthreads();
  layer_bf16<384, 128, true >(As,  W1, b1, Hs1, nullptr);
  __syncthreads();
  layer_bf16<128, 128, true >(Hs1, W2, b2, Hs2, nullptr);
  __syncthreads();
  layer_bf16<128, 128, false>(Hs2, W3, b3, nullptr, Fs);
  __syncthreads();
  for (int idx = threadIdx.x; idx < 16 * 128 / 4; idx += 256) {
    const int e = idx * 4;
    const int r = e >> 7, cc = e & 127;
    *(float4*)&c[(size_t)(row0 + r) * 128 + cc] = *(const float4*)&Fs[e];
  }
}

// --------------------------- edge scatter (segment sums) --------------------
__device__ __forceinline__ float softplus_f(float x) {
  return x > 0.f ? x + log1pf(__expf(-x)) : log1pf(__expf(x));
}

// One block (128 threads) per edge: coalesced 512B-row float atomics.
__global__ __launch_bounds__(128) void k_scatter(
    const float* __restrict__ q, const float* __restrict__ l_part,
    const float* __restrict__ c,
    const int* __restrict__ src, const int* __restrict__ dst,
    float* __restrict__ q2c, float* __restrict__ l2c,
    float* __restrict__ c2l, float sgn) {
  const int e = blockIdx.x;
  const int t = threadIdx.x;
  const size_t s = (size_t)src[e];
  const size_t d = (size_t)dst[e];
  const float sp = softplus_f(sgn * q[s * 128 + t]);
  atomicAdd(&q2c[d * 128 + t], sp);
  atomicAdd(&l2c[d * 128 + t], l_part[s * 128 + t]);
  atomicAdd(&c2l[s * 128 + t], c[d * 128 + t]);
}

// ------------- f32 row-major weights -> bf16 WMMA-B fragment-major ----------
// Fragment (kt,nt) of the KxN weight matrix lives at ((kt*NT)+nt)*512 bf16;
// within it, lane L's 16 elements (K = kt*32 + (L>>4)*16 + j, N = nt*16+(L&15))
// are contiguous at offset L*16. One thread produces one lane's 32 bytes.
__global__ __launch_bounds__(256) void k_packB(const float* __restrict__ W,
                                               bf16_t* __restrict__ out,
                                               int DIN, int DOUT) {
  const int NT = DOUT >> 4;
  const int total = (DIN >> 5) * NT * 32;
  for (int tid = blockIdx.x * 256 + threadIdx.x; tid < total; tid += gridDim.x * 256) {
    const int lane = tid & 31;
    const int frag = tid >> 5;
    const int nt = frag % NT, kt = frag / NT;
    const int n = nt * 16 + (lane & 15);
    const int kbase = kt * 32 + (lane >> 4) * 16;
    bf16_t* dst = out + ((size_t)frag << 9) + (lane << 4);
#pragma unroll
    for (int j = 0; j < 16; ++j)
      dst[j] = (bf16_t)W[(size_t)(kbase + j) * DOUT + n];
  }
}

// ============================================================================
extern "C" void kernel_launch(void* const* d_in, const int* in_sizes, int n_in,
                              void* d_out, int out_size, void* d_ws, size_t ws_size,
                              hipStream_t stream) {
  const int twoV = in_sizes[0] / 128;
  const int V    = twoV / 2;          // 50000
  const int C    = in_sizes[1] / 128; // 200000
  const float* l_in  = (const float*)d_in[0];
  const float* c_in  = (const float*)d_in[1];
  const float* noise = (const float*)d_in[2];

  // Param leaves in jax tree_flatten order (sorted keys):
  //   c_update[W0,b0,W1,b1,W2,b2], q_mlp[...], v_update[...]
  static const size_t leaf_sz[18] = {
      384 * 128, 128, 128 * 128, 128, 128 * 128, 128,   // c_update
      320 * 128, 128, 128 * 128, 128, 128 * 128, 128,   // q_mlp
      512 * 256, 256, 256 * 256, 256, 256 * 256, 256 }; // v_update
  static const int wdims[9][2] = {
      {384, 128}, {128, 128}, {128, 128},
      {320, 128}, {128, 128}, {128, 128},
      {512, 256}, {256, 256}, {256, 256} };
  const float* leaf[18];
  const int *pos_src, *pos_dst, *neg_src, *neg_dst;
  int E;
  if (n_in >= 25) {                    // params as 18 separate leaves
    for (int i = 0; i < 18; ++i) leaf[i] = (const float*)d_in[3 + i];
    pos_src = (const int*)d_in[21]; pos_dst = (const int*)d_in[22];
    neg_src = (const int*)d_in[23]; neg_dst = (const int*)d_in[24];
    E = in_sizes[21];
  } else {                             // params as one concatenated buffer
    const float* p = (const float*)d_in[3];
    for (int i = 0; i < 18; ++i) { leaf[i] = p; p += leaf_sz[i]; }
    pos_src = (const int*)d_in[4]; pos_dst = (const int*)d_in[5];
    neg_src = (const int*)d_in[6]; neg_dst = (const int*)d_in[7];
    E = in_sizes[4];
  }

  // Workspace carve-out
  size_t off = 0;
  auto carve = [&](size_t bytes) {
    void* p = (char*)d_ws + off;
    off += (bytes + 255) & ~(size_t)255;
    return p;
  };
  float* q2c  = (float*)carve((size_t)C * 128 * 4);
  float* l2c  = (float*)carve((size_t)C * 128 * 4);
  float* pc2l = (float*)carve((size_t)V * 128 * 4);
  float* nc2l = (float*)carve((size_t)V * 128 * 4);
  float* qbuf = (float*)carve((size_t)V * 128 * 4);
  static const int widx[9] = {0, 2, 4, 6, 8, 10, 12, 14, 16};
  bf16_t* wbf[9];
  for (int i = 0; i < 9; ++i) wbf[i] = (bf16_t*)carve(leaf_sz[widx[i]] * 2);

  // One-time (per call) weight swizzle to WMMA-B fragment-major bf16.
  // Total < 1 MB: L2-resident for the whole run.
  for (int i = 0; i < 9; ++i) {
    const int din = wdims[i][0], dout = wdims[i][1];
    const int total = (din >> 5) * (dout >> 4) * 32;
    k_packB<<<dim3((total + 255) / 256), dim3(256), 0, stream>>>(
        leaf[widx[i]], wbf[i], din, dout);
  }

  // State lives in d_out (in-place updates): l = d_out[0 : 2V*128], c follows.
  float* l = (float*)d_out;
  float* c = (float*)d_out + (size_t)twoV * 128;
  hipMemcpyAsync(l, l_in, (size_t)twoV * 128 * 4, hipMemcpyDeviceToDevice, stream);
  hipMemcpyAsync(c, c_in, (size_t)C * 128 * 4, hipMemcpyDeviceToDevice, stream);

  // bias pointers
  const float *cb1 = leaf[1],  *cb2 = leaf[3],  *cb3 = leaf[5];
  const float *qb1 = leaf[7],  *qb2 = leaf[9],  *qb3 = leaf[11];
  const float *vb1 = leaf[13], *vb2 = leaf[15], *vb3 = leaf[17];
  bf16_t *cW1 = wbf[0], *cW2 = wbf[1], *cW3 = wbf[2];
  bf16_t *qW1 = wbf[3], *qW2 = wbf[4], *qW3 = wbf[5];
  bf16_t *vW1 = wbf[6], *vW2 = wbf[7], *vW3 = wbf[8];

  for (int r = 0; r < 4; ++r) {
    hipMemsetAsync(q2c,  0, (size_t)C * 128 * 4, stream);
    hipMemsetAsync(l2c,  0, (size_t)C * 128 * 4, stream);
    hipMemsetAsync(pc2l, 0, (size_t)V * 128 * 4, stream);
    hipMemsetAsync(nc2l, 0, (size_t)V * 128 * 4, stream);

    const float* noise_r = noise + (size_t)r * V * 64;
    k_qmlp<<<dim3(V / 16), dim3(256), 0, stream>>>(
        l, noise_r, qW1, qb1, qW2, qb2, qW3, qb3, qbuf, V);

    k_scatter<<<dim3(E), dim3(128), 0, stream>>>(
        qbuf, l, c, pos_src, pos_dst, q2c, l2c, pc2l, +1.f);
    k_scatter<<<dim3(E), dim3(128), 0, stream>>>(
        qbuf, l + (size_t)V * 128, c, neg_src, neg_dst, q2c, l2c, nc2l, -1.f);

    k_vupdate<<<dim3(V / 16), dim3(256), 0, stream>>>(
        l, pc2l, nc2l, vW1, vb1, vW2, vb2, vW3, vb3, V);
    k_cupdate<<<dim3(C / 16), dim3(256), 0, stream>>>(
        c, l2c, q2c, cW1, cb1, cW2, cb2, cW3, cb3, C);
  }
}
